// ConvLSTM1D_1915555414732
// MI455X (gfx1250) — matrix operand order repaired and checked by reading, not compiled, not run on hardware
//
#include <hip/hip_runtime.h>

// ---------------------------------------------------------------- constants
#define BB   256
#define CC   10
#define SS   500
#define HH   256
#define NN   (BB * SS)     // 128000 rows after reshape
#define GG   (4 * HH)      // 1024 gate columns
#define GC   (3 * HH)      // 768 compact gate columns (i,g,o; f-gate unused)
#define OUTN 500
#define OUTP 512           // padded N for fc GEMM
#define K2   (SS * HH)     // 128000 (fc reduction dim)
#define KSPLIT 40
#define KCH  (K2 / KSPLIT) // 3200

typedef __attribute__((ext_vector_type(16))) __bf16 v16bf;
typedef __attribute__((ext_vector_type(8)))  __bf16 v8bf;
typedef __attribute__((ext_vector_type(8)))  float  v8f;

union ABf { v16bf v; v8bf h[2]; };

__device__ __forceinline__ unsigned short f2bf(float f) {
  unsigned int u = __float_as_uint(f);
  u += 0x7FFFu + ((u >> 16) & 1u);
  return (unsigned short)(u >> 16);
}
__device__ __forceinline__ float sigm(float x) { return 1.0f / (1.0f + __expf(-x)); }

// ---------------------------------------------------------------- bf16 casts
__global__ void cvt_w1_kernel(const float* __restrict__ w1, unsigned short* __restrict__ w1bf) {
  int i = blockIdx.x * 256 + threadIdx.x;          // GG*HH = 262144 total
  w1bf[i] = f2bf(w1[i]);
}

__global__ void cvt_fcw_kernel(const float* __restrict__ fcw, unsigned short* __restrict__ fcwbf) {
  int k = blockIdx.x * 256 + threadIdx.x;          // 0..127999
  int n = blockIdx.y;                              // 0..511 (rows 500..511 are zero pad)
  size_t idx = (size_t)n * K2 + k;
  fcwbf[idx] = (n < OUTN) ? f2bf(fcw[idx]) : (unsigned short)0;
}

// ------------------------------------------------- conv(center tap)+ReLU+LSTM0
__global__ void conv_lstm0_kernel(const float* __restrict__ x,
                                  const float* __restrict__ conv_w,
                                  const float* __restrict__ conv_b,
                                  const float* __restrict__ w_ih0,
                                  const float* __restrict__ b_ih0,
                                  const float* __restrict__ b_hh0,
                                  unsigned short* __restrict__ h0bf) {
  int n = blockIdx.x;                  // 0..127999 : (batch,timestep)
  int b = n / SS, s = n % SS;
  float y = conv_b[0];
#pragma unroll
  for (int c = 0; c < CC; ++c)
    y += x[(size_t)(b * CC + c) * SS + s] * conv_w[c * 5 + 2];  // only center tap matters
  y = fmaxf(y, 0.0f);

  int k = threadIdx.x;                 // 0..255 hidden unit
  float gi = y * w_ih0[k]          + b_ih0[k]          + b_hh0[k];
  float gg = y * w_ih0[2 * HH + k] + b_ih0[2 * HH + k] + b_hh0[2 * HH + k];
  float go = y * w_ih0[3 * HH + k] + b_ih0[3 * HH + k] + b_hh0[3 * HH + k];
  float c0 = sigm(gi) * tanhf(gg);
  float h  = sigm(go) * tanhf(c0);
  h0bf[(size_t)n * HH + k] = f2bf(h);
}

// ------------------------------------------------- LSTM1: WMMA GEMM + activation
// Block = 256 threads (8 waves). Tile: 16 rows of h0 x 768 compact gate cols.
// Wave w owns 6 N-subtiles (cols w*96 .. w*96+95 in compact space).
__global__ void lstm1_gemm_kernel(const unsigned short* __restrict__ h0bf,
                                  const unsigned short* __restrict__ w1bf,
                                  const float* __restrict__ b_ih1,
                                  const float* __restrict__ b_hh1,
                                  unsigned short* __restrict__ h1bf) {
  __shared__ unsigned short ldsA[16 * HH];   // 8 KB  : A tile bf16
  __shared__ float          ldsG[16 * GC];   // 48 KB : gate results f32

  const int tid  = threadIdx.x;
  const int wv   = tid >> 5;
  const int lane = tid & 31;
  const int lm   = lane & 15;
  const int lh   = lane >> 4;
  const int row0 = blockIdx.x * 16;

  // stage A tile: 16 rows x 256 cols bf16 (each thread copies 32 B)
  {
    int r = tid >> 4, seg = tid & 15;
    const unsigned short* src = h0bf + (size_t)(row0 + r) * HH + seg * 16;
    unsigned short* dst = &ldsA[r * HH + seg * 16];
    *(uint4*)(dst)     = *(const uint4*)(src);
    *(uint4*)(dst + 8) = *(const uint4*)(src + 8);
  }
  __syncthreads();

  v8f acc[6];
#pragma unroll
  for (int t = 0; t < 6; ++t) acc[t] = (v8f){0, 0, 0, 0, 0, 0, 0, 0};

#pragma unroll
  for (int kk = 0; kk < HH / 32; ++kk) {
    const int kb = kk * 32;
    ABf a;
    // A 16x32 bf16 fragment: lane lm = row, 16 contiguous K per lane-half
    a.h[0] = *(const v8bf*)(const void*)&ldsA[lm * HH + kb + lh * 16];
    a.h[1] = *(const v8bf*)(const void*)&ldsA[lm * HH + kb + lh * 16 + 8];
#pragma unroll
    for (int nt = 0; nt < 6; ++nt) {
      int cg = wv * 96 + nt * 16;                       // compact col base (i|g|o)
      int nb = (cg < 256 ? cg : cg + 256) + lm;         // real gate column
      // B 32x16 fragment: B[k][n] = w_ih1[n][k] -> 16 contiguous K per lane-half
      v16bf bf = *(const v16bf*)(const void*)&w1bf[(size_t)nb * HH + kb + lh * 16];
      acc[nt] = __builtin_amdgcn_wmma_f32_16x16x32_bf16(
          false, a.v, false, bf, (short)0, acc[nt], false, false);
    }
  }

  // scatter D tiles into LDS (lane lm = col, VGPR v = row v + 8*lane_half)
#pragma unroll
  for (int nt = 0; nt < 6; ++nt) {
    int cg = wv * 96 + nt * 16 + lm;
#pragma unroll
    for (int v = 0; v < 8; ++v)
      ldsG[(v + lh * 8) * GC + cg] = acc[nt][v];
  }
  __syncthreads();

  // fused activation: thread t handles hidden unit k=t for all 16 rows
  {
    int k = tid;
    float bi = b_ih1[k]          + b_hh1[k];
    float bg = b_ih1[2 * HH + k] + b_hh1[2 * HH + k];
    float bo = b_ih1[3 * HH + k] + b_hh1[3 * HH + k];
#pragma unroll
    for (int m = 0; m < 16; ++m) {
      float gi = ldsG[m * GC + k]            + bi;
      float gg = ldsG[m * GC + 256 + k]      + bg;
      float go = ldsG[m * GC + 512 + k]      + bo;
      float c  = sigm(gi) * tanhf(gg);
      float h  = sigm(go) * tanhf(c);
      h1bf[(size_t)(row0 + m) * HH + k] = f2bf(h);
    }
  }
}

// ------------------------------------------------- fc: init out with bias
__global__ void out_init_kernel(const float* __restrict__ fc_b, float* __restrict__ out) {
  int i = blockIdx.x * 256 + threadIdx.x;
  if (i < BB * OUTN) out[i] = fc_b[i % OUTN];
}

// ------------------------------------------------- fc GEMM, split-K + atomics
// Block = 256 threads (8 waves). grid = (16 Mtiles, 4, KSPLIT); wave -> N-tile.
__global__ void fc_gemm_kernel(const unsigned short* __restrict__ h1bf,
                               const unsigned short* __restrict__ fcwbf,
                               float* __restrict__ out) {
  const int tid   = threadIdx.x;
  const int wv    = tid >> 5;
  const int lane  = tid & 31;
  const int lm    = lane & 15;
  const int lh    = lane >> 4;
  const int mt    = blockIdx.x;                 // 0..15
  const int ntile = blockIdx.y * 8 + wv;        // 0..31 (cols, padded to 512)
  const int k0    = blockIdx.z * KCH;

  const unsigned short* pa = h1bf + (size_t)(mt * 16 + lm) * K2 + lh * 16;   // A row (feat)
  const unsigned short* pb = fcwbf + (size_t)(ntile * 16 + lm) * K2 + lh * 16;

  v8f acc = (v8f){0, 0, 0, 0, 0, 0, 0, 0};
  for (int k = k0; k < k0 + KCH; k += 32) {
    __builtin_prefetch(pb + k + 512, 0, 1);     // stream B ahead (global_prefetch_b8)
    ABf a;
    a.h[0] = *(const v8bf*)(const void*)(pa + k);
    a.h[1] = *(const v8bf*)(const void*)(pa + k + 8);
    v16bf bf = *(const v16bf*)(const void*)(pb + k);
    acc = __builtin_amdgcn_wmma_f32_16x16x32_bf16(
        false, a.v, false, bf, (short)0, acc, false, false);
  }

  int col = ntile * 16 + lm;
  if (col < OUTN) {
#pragma unroll
    for (int v = 0; v < 8; ++v) {
      int row = mt * 16 + v + lh * 8;
      atomicAdd(&out[row * OUTN + col], acc[v]);
    }
  }
}

// ---------------------------------------------------------------- launch
extern "C" void kernel_launch(void* const* d_in, const int* in_sizes, int n_in,
                              void* d_out, int out_size, void* d_ws, size_t ws_size,
                              hipStream_t stream) {
  const float* x      = (const float*)d_in[0];
  const float* conv_w = (const float*)d_in[1];
  const float* conv_b = (const float*)d_in[2];
  const float* w_ih0  = (const float*)d_in[3];
  const float* b_ih0  = (const float*)d_in[4];
  const float* b_hh0  = (const float*)d_in[5];
  const float* w_ih1  = (const float*)d_in[6];
  const float* b_ih1  = (const float*)d_in[7];
  const float* b_hh1  = (const float*)d_in[8];
  const float* fc_w   = (const float*)d_in[9];
  const float* fc_b   = (const float*)d_in[10];
  float* out = (float*)d_out;

  // workspace carve-up (all 256 B aligned)
  char* ws = (char*)d_ws;
  unsigned short* h0bf  = (unsigned short*)(ws);                               // 65.54 MB
  unsigned short* h1bf  = (unsigned short*)(ws + (size_t)NN * HH * 2);         // 65.54 MB
  unsigned short* w1bf  = (unsigned short*)(ws + (size_t)NN * HH * 4);         // 0.52 MB
  unsigned short* fcwbf = (unsigned short*)(ws + (size_t)NN * HH * 4 + 524288);// 131.07 MB

  // weight casts
  cvt_w1_kernel<<<GG * HH / 256, 256, 0, stream>>>(w_ih1, w1bf);
  cvt_fcw_kernel<<<dim3(K2 / 256, OUTP), 256, 0, stream>>>(fc_w, fcwbf);

  // conv + LSTM0 (elementwise)
  conv_lstm0_kernel<<<NN, 256, 0, stream>>>(x, conv_w, conv_b, w_ih0, b_ih0, b_hh0, h0bf);

  // LSTM1 GEMM + activation (WMMA bf16)
  lstm1_gemm_kernel<<<NN / 16, 256, 0, stream>>>(h0bf, w1bf, b_ih1, b_hh1, h1bf);

  // fc: bias init, then split-K WMMA GEMM with f32 atomic reduction
  out_init_kernel<<<(BB * OUTN + 255) / 256, 256, 0, stream>>>(fc_b, out);
  fc_gemm_kernel<<<dim3(16, 4, KSPLIT), 256, 0, stream>>>(h1bf, fcwbf, out);
}
